// Attention_38173669327601
// MI455X (gfx1250) — compile-verified, matrix operand order checked by reading, not run on hardware
//
#include <hip/hip_runtime.h>
#include <hip/hip_bf16.h>

typedef __attribute__((ext_vector_type(16))) __bf16 v16bf;
typedef __attribute__((ext_vector_type(8)))  __bf16 v8bf;
typedef __attribute__((ext_vector_type(8)))  float  v8f;
typedef __attribute__((ext_vector_type(4)))  float  v4f;
typedef __attribute__((ext_vector_type(2)))  float  v2f;

#define M_LEN  1024
#define BATCH  16384
#define KCH    32              // K-chunk per WMMA pass (bf16 K=32)
#define NCHUNK (M_LEN / KCH)   // 32

// ---------------------------------------------------------------------------
// Kernel 0: convert W (fp32, row-major [k][l]) -> bf16 in workspace.
// B fragments need W rows contiguous along l == existing layout, no transpose.
// ---------------------------------------------------------------------------
__global__ void cvt_w_bf16(const float* __restrict__ W, __bf16* __restrict__ Wb) {
    int i = (blockIdx.x * 256 + threadIdx.x) * 4;
    v4f w = *(const v4f*)(W + i);
    Wb[i + 0] = (__bf16)w[0];
    Wb[i + 1] = (__bf16)w[1];
    Wb[i + 2] = (__bf16)w[2];
    Wb[i + 3] = (__bf16)w[3];
}

// ---------------------------------------------------------------------------
// Kernel 1: fused scores(GEMM bf16->f32) + softmax(over 4 modalities) +
//           weighted sum + zero-modality rescale.
// Grid: BATCH/16 WGs x 1024 threads (32 waves). Each WG owns a 16-row batch
// tile and all 1024 output columns; each wave owns 2 column tiles of 16.
// Per chunk (unroll disabled to cap VGPR pressure):
//   1) issue B-fragment global loads (latency hidden behind barrier+convert)
//   2) s_wait_asynccnt + barrier  (chunk-i fp32 DMA visible everywhere)
//   3) issue async DMA for chunk i+1 (overlaps everything below)
//   4) 1024-thread fp32->bf16 convert of the 4 KB tile, barrier
//   5) bf16 A-fragment ds_loads streamed per modality + 8 WMMAs
// ---------------------------------------------------------------------------
__global__ __launch_bounds__(1024, 1)
void fused_modal_attn(const float* __restrict__ x0, const float* __restrict__ x1,
                      const float* __restrict__ x2, const float* __restrict__ x3,
                      const __bf16* __restrict__ Wb, float* __restrict__ out) {
    __shared__ __align__(16) float  xsf[2][4][16][KCH];  // 16 KB fp32 DMA buffers
    __shared__ __align__(16) __bf16 xbh[4][16][KCH];     // 4 KB bf16 compute tile
    __shared__ float rowsum[4][16];
    __shared__ float scaler[16];

    const int tid     = threadIdx.x;
    const int lane    = tid & 31;
    const int wave    = tid >> 5;          // 0..31
    const int l16     = lane & 15;
    const int halfsel = lane >> 4;         // 0 or 1
    const int ko      = halfsel * 8;       // per-lane K offset within chunk
    const int b0      = blockIdx.x * 16;   // batch-row tile base

    // ---- per-thread staging coordinates (resolved ONCE, no per-iter select)
    // each thread DMAs / converts 2 consecutive floats per chunk
    const int e  = tid * 2;                // 0..2046
    const int sm = e >> 9;                 // modality 0..3
    const int sr = (e >> 5) & 15;          // row in tile
    const int sc = e & 31;                 // col in chunk (even)
    const float* gp =
        ((sm == 0) ? x0 : (sm == 1) ? x1 : (sm == 2) ? x2 : x3) +
        (size_t)(b0 + sr) * M_LEN + sc;
    const unsigned ldsA = (unsigned)(uintptr_t)&xsf[0][sm][sr][sc];
    const unsigned ldsB = (unsigned)(uintptr_t)&xsf[1][sm][sr][sc];

    // per-lane B base pointers (hoisted out of the k-loop)
    const __bf16* wpb0 = Wb + (size_t)((wave * 2 + 0) * 16 + l16) * M_LEN + ko;
    const __bf16* wpb1 = Wb + (size_t)((wave * 2 + 1) * 16 + l16) * M_LEN + ko;

    // kick off chunk-0 DMA; it overlaps the row-sum phase below
    asm volatile("global_load_async_to_lds_b64 %0, %1, off"
                 :: "v"(ldsA), "v"(gp) : "memory");

    // ---- Phase 0: per-(modality,row) sums -> zero-count scaler -------------
    if (tid < 64) rowsum[tid >> 4][tid & 15] = 0.0f;
    __syncthreads();
    {
        // 1024 threads = 4 modalities * 16 rows * 16 threads/row * 64 elems
        int m  = tid >> 8;
        int r  = (tid >> 4) & 15;
        int c0 = (tid & 15) * 64;
        const float* p = (m == 0) ? x0 : (m == 1) ? x1 : (m == 2) ? x2 : x3;
        p += (size_t)(b0 + r) * M_LEN + c0;
        float s = 0.0f;
        #pragma unroll 8
        for (int i = 0; i < 64; ++i) s += p[i];
        atomicAdd(&rowsum[m][r], s);     // ds_add_f32
    }
    __syncthreads();
    if (tid < 16) {
        int z = 0;
        #pragma unroll
        for (int m = 0; m < 4; ++m) z += (rowsum[m][tid] == 0.0f) ? 1 : 0;
        scaler[tid] = (z > 0) ? (float)(z + 1) : 1.0f;
    }
    // (the k-loop's first barrier separates this from LDS reuse)

    // ---- Phase 1: 4-modality GEMM, f32 accumulators in registers ----------
    v8f acc[2][4];
    #pragma unroll
    for (int t = 0; t < 2; ++t)
        #pragma unroll
        for (int m = 0; m < 4; ++m)
            #pragma unroll
            for (int i = 0; i < 8; ++i) acc[t][m][i] = 0.0f;

    #pragma clang loop unroll(disable)
    for (int ic = 0; ic < NCHUNK; ++ic) {
        const int kk = ic * KCH;

        // (1) B fragments for THIS chunk: pure global loads, no LDS
        // dependency -> issue first so latency hides behind barrier+convert.
        // B[l][n] = W[n][l]; lane owns column n.
        union BU { v16bf v; v8bf h[2]; };
        BU ub0, ub1;
        {
            const __bf16* w0 = wpb0 + kk;
            const __bf16* w1 = wpb1 + kk;
            ub0.h[0] = *(const v8bf*)(w0);
            ub0.h[1] = *(const v8bf*)(w0 + 16);
            ub1.h[0] = *(const v8bf*)(w1);
            ub1.h[1] = *(const v8bf*)(w1 + 16);
        }

        // (2) my chunk-ic DMA complete; barrier => everyone's complete, and
        // all reads of xbh / the other fp32 buffer have retired
        asm volatile("s_wait_asynccnt 0x0" ::: "memory");
        __syncthreads();

        // (3) prefetch chunk ic+1 into the other fp32 buffer
        if (ic + 1 < NCHUNK) {
            const float* gn = gp + (size_t)(ic + 1) * KCH;
            unsigned ld = ((ic + 1) & 1) ? ldsB : ldsA;
            asm volatile("global_load_async_to_lds_b64 %0, %1, off"
                         :: "v"(ld), "v"(gn) : "memory");
        }

        // (4) convert step: 1024 threads turn the 8 KB fp32 tile into bf16
        // (one ds_load_b64 + v_cvt_pk_bf16_f32 + ds_store_b32 each)
        {
            v2f w = *(const v2f*)&xsf[ic & 1][sm][sr][sc];
            xbh[sm][sr][sc]     = (__bf16)w[0];
            xbh[sm][sr][sc + 1] = (__bf16)w[1];
        }
        __syncthreads();

        // (5) stream A fragments one modality at a time (16-bit A 16x32
        // layout: lane<16 -> K {0..7,16..23}, lane>=16 -> K {8..15,24..31};
        // row M = lane%16): 2 ds_load_b128 each, then both tile WMMAs.
        #pragma unroll
        for (int m = 0; m < 4; ++m) {
            union { v16bf v; v8bf h[2]; } ua;
            ua.h[0] = *(const v8bf*)&xbh[m][l16][ko];
            ua.h[1] = *(const v8bf*)&xbh[m][l16][ko + 16];
            acc[0][m] = __builtin_amdgcn_wmma_f32_16x16x32_bf16(
                false, ua.v, false, ub0.v, (short)0, acc[0][m], false, false);
            acc[1][m] = __builtin_amdgcn_wmma_f32_16x16x32_bf16(
                false, ua.v, false, ub1.v, (short)0, acc[1][m], false, false);
        }
    }

    // ---- Phase 2: softmax over modalities + fused sum + rescale -----------
    // D layout: VGPR i -> row (i + 8*halfsel), col = n0 + lane%16
    #pragma unroll
    for (int t = 0; t < 2; ++t) {
        int col   = (wave * 2 + t) * 16 + l16;
        int rbase = b0 + halfsel * 8;
        #pragma unroll
        for (int i = 0; i < 8; ++i) {
            float s0 = acc[t][0][i], s1 = acc[t][1][i];
            float s2 = acc[t][2][i], s3 = acc[t][3][i];
            float mx = fmaxf(fmaxf(s0, s1), fmaxf(s2, s3));
            float e0 = __expf(s0 - mx), e1 = __expf(s1 - mx);
            float e2 = __expf(s2 - mx), e3 = __expf(s3 - mx);
            float inv = __frcp_rn(e0 + e1 + e2 + e3);
            size_t off = (size_t)(rbase + i) * M_LEN + col;
            float f = (x0[off] * e0 + x1[off] * e1 +
                       x2[off] * e2 + x3[off] * e3) * inv;
            out[off] = f * scaler[i + halfsel * 8];
        }
    }
}

// ---------------------------------------------------------------------------
extern "C" void kernel_launch(void* const* d_in, const int* in_sizes, int n_in,
                              void* d_out, int out_size, void* d_ws, size_t ws_size,
                              hipStream_t stream) {
    const float* x0 = (const float*)d_in[0];
    const float* x1 = (const float*)d_in[1];
    const float* x2 = (const float*)d_in[2];
    const float* x3 = (const float*)d_in[3];
    const float* W  = (const float*)d_in[4];
    __bf16* Wb = (__bf16*)d_ws;          // 1024*1024*2 = 2 MB scratch

    cvt_w_bf16<<<(M_LEN * M_LEN) / (256 * 4), 256, 0, stream>>>(W, Wb);
    fused_modal_attn<<<BATCH / 16, 1024, 0, stream>>>(x0, x1, x2, x3, Wb,
                                                      (float*)d_out);
}
